// Sage_69630009802903
// MI455X (gfx1250) — matrix operand order repaired and checked by reading, not compiled, not run on hardware
//
#include <hip/hip_runtime.h>
#include <hip/hip_bf16.h>

typedef __bf16 bf16_t;
typedef bf16_t v16bf __attribute__((ext_vector_type(16)));
typedef bf16_t v8bf  __attribute__((ext_vector_type(8)));
typedef float  v8f   __attribute__((ext_vector_type(8)));

// ---- f32 -> bf16 helpers ----

// accurate RNE (used only in tiny weight-pack kernel)
static __device__ __forceinline__ bf16_t f2bf(float f) {
    union { float f; unsigned u; } in; in.f = f;
    unsigned u = in.u;
    unsigned r = (u + 0x7FFFu + ((u >> 16) & 1u)) >> 16;
    unsigned short s = (unsigned short)r;
    bf16_t b;
    __builtin_memcpy(&b, &s, sizeof(s));
    return b;
}

// fast path: round-half-up + v_perm_b32 pack of two floats into bf16x2 (one dword)
static __device__ __forceinline__ unsigned pk2bf(float lo, float hi) {
    unsigned ul = __float_as_uint(lo) + 0x8000u;
    unsigned uh = __float_as_uint(hi) + 0x8000u;
    // dst bytes: [1:0] = ul[3:2], [3:2] = uh[3:2]  (src1 supplies bytes 0..3)
    return __builtin_amdgcn_perm(uh, ul, 0x07060302u);
}

// ---------------- utility kernels ----------------

__global__ void zero_f32(float* __restrict__ p, int n) {
    int i = blockIdx.x * blockDim.x + threadIdx.x;
    if (i < n) p[i] = 0.0f;
}

// Pack [K x ncol_l] Wl and [K x ncol_r] Wr into transposed bf16 WT[ncol_total][K]
__global__ void pack_wT(const float* __restrict__ Wl, const float* __restrict__ Wr,
                        bf16_t* __restrict__ WT, int K, int ncol_l, int ncol_total) {
    int t = blockIdx.x * blockDim.x + threadIdx.x;
    if (t >= K * ncol_total) return;
    int n = t / K;
    int k = t - n * K;
    int ncol_r = ncol_total - ncol_l;
    float v = (n < ncol_l) ? Wl[(size_t)k * ncol_l + n]
                           : Wr[(size_t)k * ncol_r + (n - ncol_l)];
    WT[(size_t)n * K + k] = f2bf(v);
}

__global__ void degree_kernel(const int* __restrict__ dst, int nE, float* __restrict__ cnt) {
    int e = blockIdx.x * blockDim.x + threadIdx.x;
    if (e < nE) atomicAdd(&cnt[dst[e]], 1.0f);
}

// ---------------- WMMA GEMM: C[M][NCOL] = A[M][K] x BT[NCOL][K](bf16) ----------------
// AT = float (converted on the fly) or bf16_t (direct fragment loads).
// One wave per 16x16 output tile; wave w in block handles column tile w.
// blockDim.x = 32 * (NCOL/16); gridDim.x = ceil(M/16).
template <int K, typename AT>
__global__ void gemm_wmma_bf16(const AT* __restrict__ A, const bf16_t* __restrict__ BT,
                               float* __restrict__ C, int M, int NCOL) {
    const int wave = threadIdx.x >> 5;
    const int lane = threadIdx.x & 31;
    const int half = lane >> 4;    // K-half selector
    const int r    = lane & 15;    // row (A) / col (B,C) within tile
    const int m0   = blockIdx.x * 16;
    const int ncol = wave * 16 + r;

    const int arow_i = min(m0 + r, M - 1);
    const AT*     __restrict__ arow = A  + (size_t)arow_i * K;
    const bf16_t* __restrict__ brow = BT + (size_t)ncol   * K;

    v8f acc = {0.f, 0.f, 0.f, 0.f, 0.f, 0.f, 0.f, 0.f};

#pragma unroll
    for (int k0 = 0; k0 < K; k0 += 32) {
        // A fragment (16-bit A layout): lane holds K = k0+8h..+7 and k0+16+8h..+7
        v16bf a;
        if constexpr (sizeof(AT) == 4) {
            const float4 x0 = *(const float4*)(arow + k0 + 8 * half);
            const float4 x1 = *(const float4*)(arow + k0 + 8 * half + 4);
            const float4 x2 = *(const float4*)(arow + k0 + 16 + 8 * half);
            const float4 x3 = *(const float4*)(arow + k0 + 16 + 8 * half + 4);
            union { unsigned u[8]; v16bf v; } au;
            au.u[0] = pk2bf(x0.x, x0.y); au.u[1] = pk2bf(x0.z, x0.w);
            au.u[2] = pk2bf(x1.x, x1.y); au.u[3] = pk2bf(x1.z, x1.w);
            au.u[4] = pk2bf(x2.x, x2.y); au.u[5] = pk2bf(x2.z, x2.w);
            au.u[6] = pk2bf(x3.x, x3.y); au.u[7] = pk2bf(x3.z, x3.w);
            a = au.v;
        } else {
            v8bf lo = *(const v8bf*)(arow + k0 + 8 * half);
            v8bf hi = *(const v8bf*)(arow + k0 + 16 + 8 * half);
            a = __builtin_shufflevector(lo, hi, 0, 1, 2, 3, 4, 5, 6, 7,
                                        8, 9, 10, 11, 12, 13, 14, 15);
        }
        // B fragment: lane holds column ncol, K = k0+8h..+7 and k0+16+8h..+7
        v8bf blo = *(const v8bf*)(brow + k0 + 8 * half);
        v8bf bhi = *(const v8bf*)(brow + k0 + 16 + 8 * half);
        v16bf b = __builtin_shufflevector(blo, bhi, 0, 1, 2, 3, 4, 5, 6, 7,
                                          8, 9, 10, 11, 12, 13, 14, 15);
        acc = __builtin_amdgcn_wmma_f32_16x16x32_bf16(false, a, false, b,
                                                      (short)0, acc, false, false);
    }

#pragma unroll
    for (int rr = 0; rr < 8; ++rr) {
        int row = m0 + rr + 8 * half;   // C layout: VGPR rr -> row rr + 8*half, col = r
        if (row < M) C[(size_t)row * NCOL + ncol] = acc[rr];
    }
}

// ---------------- edge scatter-add of projected features ----------------
// F4 lanes per edge, float4 per lane. agg[dst] += P[src][coff : coff+F]
__global__ void edge_agg(const float* __restrict__ P, int ldP, int coff,
                         const int* __restrict__ src, const int* __restrict__ dst,
                         float* __restrict__ agg, int F4, int F, int total) {
    int t = blockIdx.x * blockDim.x + threadIdx.x;
    if (t >= total) return;
    int e  = t / F4;
    int c4 = (t - e * F4) * 4;
    int s = src[e];
    int d = dst[e];
    const float4 v = *(const float4*)(P + (size_t)s * ldP + coff + c4);
    float* o = agg + (size_t)d * F + c4;
    atomicAdd(o + 0, v.x);
    atomicAdd(o + 1, v.y);
    atomicAdd(o + 2, v.z);
    atomicAdd(o + 3, v.w);
}

// ---------------- hb = bf16(relu(P1_l + agg/cnt + b1)), F=64 ----------------
__global__ void combine_relu(const float* __restrict__ P1, const float* __restrict__ agg,
                             const float* __restrict__ cnt, const float* __restrict__ bias,
                             bf16_t* __restrict__ hb, int N) {
    int t = blockIdx.x * blockDim.x + threadIdx.x;
    int i  = t >> 4;
    int c4 = (t & 15) * 4;
    if (i >= N) return;
    float inv = 1.0f / fmaxf(cnt[i], 1.0f);
    const float4 l = *(const float4*)(P1  + (size_t)i * 128 + c4);
    const float4 a = *(const float4*)(agg + (size_t)i * 64  + c4);
    const float4 b = *(const float4*)(bias + c4);
    float4 o;
    o.x = fmaxf(l.x + a.x * inv + b.x, 0.0f);
    o.y = fmaxf(l.y + a.y * inv + b.y, 0.0f);
    o.z = fmaxf(l.z + a.z * inv + b.z, 0.0f);
    o.w = fmaxf(l.w + a.w * inv + b.w, 0.0f);
    uint2 st;
    st.x = pk2bf(o.x, o.y);
    st.y = pk2bf(o.z, o.w);
    *(uint2*)((unsigned*)hb + (size_t)i * 32 + (c4 >> 1)) = st;
}

// ---------------- out = softmax(P2_l + agg/cnt + b2), 40 classes ----------------
__global__ void combine_softmax(const float* __restrict__ P2, const float* __restrict__ agg,
                                const float* __restrict__ cnt, const float* __restrict__ bias,
                                float* __restrict__ out, int N) {
    int i = blockIdx.x * blockDim.x + threadIdx.x;
    if (i >= N) return;
    float inv = 1.0f / fmaxf(cnt[i], 1.0f);
    float v[40];
    float mx = -3.402823466e38f;
#pragma unroll
    for (int c = 0; c < 40; ++c) {
        float t = P2[(size_t)i * 80 + c] + agg[(size_t)i * 40 + c] * inv + bias[c];
        v[c] = t;
        mx = fmaxf(mx, t);
    }
    float s = 0.0f;
#pragma unroll
    for (int c = 0; c < 40; ++c) {
        float e = __expf(v[c] - mx);
        v[c] = e;
        s += e;
    }
    float is = 1.0f / s;
#pragma unroll
    for (int c = 0; c < 40; ++c) out[(size_t)i * 40 + c] = v[c] * is;
}

// ---------------- host side ----------------

extern "C" void kernel_launch(void* const* d_in, const int* in_sizes, int n_in,
                              void* d_out, int out_size, void* d_ws, size_t ws_size,
                              hipStream_t stream) {
    const float* x   = (const float*)d_in[0];
    const int*   ei  = (const int*)d_in[1];
    const float* W1l = (const float*)d_in[2];
    const float* W1r = (const float*)d_in[3];
    const float* b1  = (const float*)d_in[4];
    const float* W2l = (const float*)d_in[5];
    const float* W2r = (const float*)d_in[6];
    const float* b2  = (const float*)d_in[7];
    float* out = (float*)d_out;

    const int N = in_sizes[0] / 256;   // 100000
    const int E = in_sizes[1] / 2;     // 1600000
    const int* src = ei;
    const int* dst = ei + E;

    // workspace layout (512B-aligned regions)
    char* ws = (char*)d_ws;
    size_t off = 0;
    auto alloc = [&](size_t bytes) -> char* {
        char* p = ws + off;
        off += (bytes + 511) & ~(size_t)511;
        return p;
    };
    float*  P1   = (float*)alloc((size_t)N * 128 * sizeof(float)); // [x@W1l | x@W1r]
    float*  P2   = (float*)alloc((size_t)N * 80  * sizeof(float)); // [h@W2l | h@W2r]
    bf16_t* hb   = (bf16_t*)alloc((size_t)N * 64 * sizeof(bf16_t));
    float*  agg1 = (float*)alloc((size_t)N * 64  * sizeof(float));
    float*  agg2 = (float*)alloc((size_t)N * 40  * sizeof(float));
    float*  cnt  = (float*)alloc((size_t)N * sizeof(float));
    bf16_t* WT1  = (bf16_t*)alloc((size_t)128 * 256 * sizeof(bf16_t));
    bf16_t* WT2  = (bf16_t*)alloc((size_t)80  * 64  * sizeof(bf16_t));
    (void)ws_size; (void)n_in; (void)out_size;

    const int B = 256;

    // 1) zero accumulators (ws is not re-poisoned between replays)
    {
        int n1 = N * 64, n2 = N * 40;
        zero_f32<<<(n1 + B - 1) / B, B, 0, stream>>>(agg1, n1);
        zero_f32<<<(n2 + B - 1) / B, B, 0, stream>>>(agg2, n2);
        zero_f32<<<(N + B - 1) / B, B, 0, stream>>>(cnt, N);
    }

    // 2) pack weights to transposed bf16
    pack_wT<<<(256 * 128 + B - 1) / B, B, 0, stream>>>(W1l, W1r, WT1, 256, 64, 128);
    pack_wT<<<(64 * 80 + B - 1) / B, B, 0, stream>>>(W2l, W2r, WT2, 64, 40, 80);

    // 3) in-degree counts
    degree_kernel<<<(E + B - 1) / B, B, 0, stream>>>(dst, E, cnt);

    // 4) layer-1 GEMM: P1 = x @ [W1l|W1r]   (M=N, K=256, NCOL=128 -> 8 waves/block)
    gemm_wmma_bf16<256, float><<<(N + 15) / 16, 8 * 32, 0, stream>>>(x, WT1, P1, N, 128);

    // 5) scatter-add projected neighbor features (cols 64..127 of P1)
    edge_agg<<<(E * 16 + B - 1) / B, B, 0, stream>>>(P1, 128, 64, src, dst, agg1, 16, 64, E * 16);

    // 6) hb = bf16(relu(P1_l + agg1/cnt + b1))
    combine_relu<<<(N * 16 + B - 1) / B, B, 0, stream>>>(P1, agg1, cnt, b1, hb, N);

    // 7) layer-2 GEMM: P2 = hb @ [W2l|W2r]  (K=64, NCOL=80 -> 5 waves/block)
    gemm_wmma_bf16<64, bf16_t><<<(N + 15) / 16, 5 * 32, 0, stream>>>(hb, WT2, P2, N, 80);

    // 8) scatter-add projected neighbor features (cols 40..79 of P2)
    edge_agg<<<(E * 10 + B - 1) / B, B, 0, stream>>>(P2, 80, 40, src, dst, agg2, 10, 40, E * 10);

    // 9) softmax
    combine_softmax<<<(N + B - 1) / B, B, 0, stream>>>(P2, agg2, cnt, b2, out, N);
}
